// MuiInt8GateUpDownMLP_51874615001498
// MI455X (gfx1250) — matrix operand order repaired and checked by reading, not compiled, not run on hardware
//
#include <hip/hip_runtime.h>
#include <hip/hip_bf16.h>
#include <stdint.h>

#define HIDDEN 4096
#define INTER  11008
#define TOKENS 128
#define APAD   136   // LDS row pitch in halves (128 data + 8 pad, conflict-free b128)

typedef __attribute__((ext_vector_type(16))) _Float16 v16h;
typedef __attribute__((ext_vector_type(8)))  _Float16 v8h;
typedef __attribute__((ext_vector_type(2)))  _Float16 h2;
typedef __attribute__((ext_vector_type(8)))  float    v8f;

__device__ __forceinline__ h2 splat_h2(float v) {
  _Float16 h = (_Float16)v;
  h2 r; r[0] = h; r[1] = h;
  return r;
}

// q in [0,256): f16 bits (0x6400|q) == value (1024+q) exactly.
// w = s*q + m == s*(1024+q) + (m - 1024*s)  -> v_lshl + v_or3 + v_pk_fma_f16
__device__ __forceinline__ uint32_t deq_pair(uint32_t qlo, uint32_t qhi, h2 s, h2 madj) {
  uint32_t bits = 0x64006400u | qlo | (qhi << 16);
  h2 w = __builtin_bit_cast(h2, bits);
  h2 r = w * s + madj;            // v_pk_fma_f16
  return __builtin_bit_cast(uint32_t, r);
}

__device__ __forceinline__ v8f wmma_f16(const v16h& a, const v16h& b, v8f c) {
  return __builtin_amdgcn_wmma_f32_16x16x32_f16(false, a, false, b,
                                                (short)0, c, false, false);
}

// Load one A fragment (16x32 f16 WMMA layout) for this lane from LDS
__device__ __forceinline__ v16h load_a_frag(const _Float16* ar, int half) {
  v8h alo = *(const v8h*)(ar + 16 * half);
  v8h ahi = *(const v8h*)(ar + 16 * half + 8);
  v16h a;
#pragma unroll
  for (int j = 0; j < 8; ++j) { a[j] = alo[j]; a[8 + j] = ahi[j]; }
  return a;
}

// ---------------------------------------------------------------------------
// Kernel 1: RMSNorm -> f16 activations
// ---------------------------------------------------------------------------
__global__ __launch_bounds__(256) void rmsnorm_f16_kernel(
    const float* __restrict__ inp, const float* __restrict__ nw,
    _Float16* __restrict__ xh) {
  const int t = blockIdx.x;
  const float* x = inp + (size_t)t * HIDDEN;
  __shared__ float red[256];
  float ss = 0.f;
  for (int k = threadIdx.x; k < HIDDEN; k += 256) { float v = x[k]; ss += v * v; }
  red[threadIdx.x] = ss;
  __syncthreads();
  for (int s = 128; s > 0; s >>= 1) {
    if ((int)threadIdx.x < s) red[threadIdx.x] += red[threadIdx.x + s];
    __syncthreads();
  }
  const float rinv = rsqrtf(red[0] * (1.0f / HIDDEN) + 1e-6f);
  for (int k = threadIdx.x; k < HIDDEN; k += 256) {
    xh[(size_t)t * HIDDEN + k] = (_Float16)(x[k] * rinv * nw[k]);
  }
}

// ---------------------------------------------------------------------------
// Kernel 2: gate/up GEMM + SiLU*up.  Grid (INTER/128, 2), 256 thr = 8 waves.
// Wave: N=16 inter rows, M=64 tokens (4 m-tiles of accumulators).
// A staged per 128-K group in LDS; B dequantized once, reused by 4 m-tiles.
// All 4 A fragments preloaded -> one ds wait, 8 back-to-back wmma per kc.
// ---------------------------------------------------------------------------
__global__ __launch_bounds__(256) void gateup_wmma_kernel(
    const _Float16* __restrict__ xh, const int* __restrict__ gq,
    const float* __restrict__ gsm, _Float16* __restrict__ hbuf) {
  __shared__ _Float16 lA[64 * APAD];
  const int tid  = threadIdx.x;
  const int wave = tid >> 5, lane = tid & 31;
  const int half = lane >> 4, mn = lane & 15;
  const int n0    = blockIdx.x * 128 + wave * 16;
  const int mbase = blockIdx.y * 64;

  const int    row   = n0 + mn;                       // inter row (weights)
  const int*   qrow  = gq + (size_t)row * HIDDEN * 2; // interleaved gate/up
  const float* smrow = gsm + (size_t)row * 128;       // [2][32][2]

  v8f accg[4] = {{0}}, accu[4] = {{0}};

  // staging: 4 threads per token row, 32 halves (64B) each
  const int srow = tid >> 2;
  const int scol = (tid & 3) * 32;
  const _Float16* gsrc = xh + (size_t)(mbase + srow) * HIDDEN + scol;
  _Float16* sdst = lA + srow * APAD + scol;

  for (int g = 0; g < HIDDEN / 128; ++g) {
    __syncthreads();
    {
      const v8h* src = (const v8h*)(gsrc + g * 128);
#pragma unroll
      for (int j = 0; j < 4; ++j) *(v8h*)(sdst + j * 8) = src[j];
    }
    __syncthreads();

    const float sgf = smrow[g * 2],      mgf = smrow[g * 2 + 1];
    const float suf = smrow[64 + g * 2], muf = smrow[64 + g * 2 + 1];
    const h2 sg = splat_h2(sgf), mg = splat_h2(mgf - 1024.f * sgf);
    const h2 su = splat_h2(suf), mu = splat_h2(muf - 1024.f * suf);

#pragma unroll
    for (int kc = 0; kc < 4; ++kc) {
      // ---- dequant B (gate & up), lane holds K = kb..kb+15
      const int kb = g * 128 + kc * 32 + 16 * half;
      const int4* pq = (const int4*)(qrow + (size_t)kb * 2);
      __builtin_prefetch(qrow + (size_t)(kb + 32) * 2, 0, 1);
      union { uint32_t u[8]; v16h v; } BG, BU;
#pragma unroll
      for (int j = 0; j < 8; ++j) {
        int4 q = pq[j];  // {gate k, up k, gate k+1, up k+1}
        BG.u[j] = deq_pair((uint32_t)q.x, (uint32_t)q.z, sg, mg);
        BU.u[j] = deq_pair((uint32_t)q.y, (uint32_t)q.w, su, mu);
      }
      // ---- preload all 4 A fragments (one ds clause, one wait)
      v16h a[4];
#pragma unroll
      for (int mt = 0; mt < 4; ++mt)
        a[mt] = load_a_frag(lA + (mt * 16 + mn) * APAD + kc * 32, half);
      // ---- 8 independent wmma back-to-back
#pragma unroll
      for (int mt = 0; mt < 4; ++mt) {
        accg[mt] = wmma_f16(a[mt], BG.v, accg[mt]);
        accu[mt] = wmma_f16(a[mt], BU.v, accu[mt]);
      }
    }
  }

  // epilogue: D layout M = r + 8*half, N = mn;  h = silu(g)*u
#pragma unroll
  for (int mt = 0; mt < 4; ++mt) {
#pragma unroll
    for (int r = 0; r < 8; ++r) {
      const int m = mbase + mt * 16 + r + 8 * half;
      const float gv = accg[mt][r];
      const float uv = accu[mt][r];
      const float hv = gv * __builtin_amdgcn_rcpf(1.0f + __expf(-gv)) * uv;
      hbuf[(size_t)m * INTER + (n0 + mn)] = (_Float16)hv;
    }
  }
}

// ---------------------------------------------------------------------------
// Kernel 3: down GEMM + residual.  Grid (HIDDEN/128, 2), 256 thr = 8 waves.
// ---------------------------------------------------------------------------
__global__ __launch_bounds__(256) void down_wmma_kernel(
    const _Float16* __restrict__ hbuf, const int* __restrict__ dq,
    const float* __restrict__ dsm, const float* __restrict__ residual,
    float* __restrict__ out) {
  __shared__ _Float16 lA[64 * APAD];
  const int tid  = threadIdx.x;
  const int wave = tid >> 5, lane = tid & 31;
  const int half = lane >> 4, mn = lane & 15;
  const int n0    = blockIdx.x * 128 + wave * 16;   // hidden cols
  const int mbase = blockIdx.y * 64;                // token half

  const int    row   = n0 + mn;
  const int*   qrow  = dq + (size_t)row * INTER;
  const float* smrow = dsm + (size_t)row * (2 * (INTER / 128));  // 172 floats

  v8f acc[4] = {{0}};

  const int srow = tid >> 2;
  const int scol = (tid & 3) * 32;
  const _Float16* gsrc = hbuf + (size_t)(mbase + srow) * INTER + scol;
  _Float16* sdst = lA + srow * APAD + scol;

  for (int g = 0; g < INTER / 128; ++g) {
    __syncthreads();
    {
      const v8h* src = (const v8h*)(gsrc + g * 128);
#pragma unroll
      for (int j = 0; j < 4; ++j) *(v8h*)(sdst + j * 8) = src[j];
    }
    __syncthreads();

    const float sf = smrow[g * 2], mf = smrow[g * 2 + 1];
    const h2 s = splat_h2(sf), ma = splat_h2(mf - 1024.f * sf);

#pragma unroll
    for (int kc = 0; kc < 4; ++kc) {
      const int kb = g * 128 + kc * 32 + 16 * half;
      const int4* pq = (const int4*)(qrow + (size_t)kb);
      __builtin_prefetch(qrow + (size_t)(kb + 32), 0, 1);
      union { uint32_t u[8]; v16h v; } B;
#pragma unroll
      for (int j = 0; j < 4; ++j) {
        int4 q = pq[j];  // 4 consecutive k weights
        B.u[2 * j]     = deq_pair((uint32_t)q.x, (uint32_t)q.y, s, ma);
        B.u[2 * j + 1] = deq_pair((uint32_t)q.z, (uint32_t)q.w, s, ma);
      }
      v16h a[4];
#pragma unroll
      for (int mt = 0; mt < 4; ++mt)
        a[mt] = load_a_frag(lA + (mt * 16 + mn) * APAD + kc * 32, half);
#pragma unroll
      for (int mt = 0; mt < 4; ++mt)
        acc[mt] = wmma_f16(a[mt], B.v, acc[mt]);
    }
  }

#pragma unroll
  for (int mt = 0; mt < 4; ++mt) {
#pragma unroll
    for (int r = 0; r < 8; ++r) {
      const int m = mbase + mt * 16 + r + 8 * half;
      const size_t idx = (size_t)m * HIDDEN + (n0 + mn);
      out[idx] = acc[mt][r] + residual[idx];
    }
  }
}

// ---------------------------------------------------------------------------
extern "C" void kernel_launch(void* const* d_in, const int* in_sizes, int n_in,
                              void* d_out, int out_size, void* d_ws, size_t ws_size,
                              hipStream_t stream) {
  (void)in_sizes; (void)n_in; (void)out_size; (void)ws_size;
  const float* inp      = (const float*)d_in[0];
  const float* residual = (const float*)d_in[1];
  const float* nw       = (const float*)d_in[2];
  const int*   gq       = (const int*)d_in[3];
  const float* gsm      = (const float*)d_in[4];
  const int*   dq       = (const int*)d_in[5];
  const float* dsm      = (const float*)d_in[6];
  float* out = (float*)d_out;

  _Float16* xh   = (_Float16*)d_ws;
  _Float16* hbuf = (_Float16*)((char*)d_ws + (size_t)TOKENS * HIDDEN * sizeof(_Float16));

  rmsnorm_f16_kernel<<<TOKENS, 256, 0, stream>>>(inp, nw, xh);

  dim3 g2(INTER / 128, 2);   // (86, 2)
  gateup_wmma_kernel<<<g2, 256, 0, stream>>>(xh, gq, gsm, hbuf);

  dim3 g3(HIDDEN / 128, 2);  // (32, 2)
  down_wmma_kernel<<<g3, 256, 0, stream>>>(hbuf, dq, dsm, residual, out);
}
